// MultiHeadAttention_22411139350811
// MI455X (gfx1250) — matrix operand order repaired and checked by reading, not compiled, not run on hardware
//
#include <hip/hip_runtime.h>

#define D_MODEL 512
#define N_HEADS 8
#define HEAD_DIM 64
#define S_LEN 4096
#define BATCH 2
#define M_TOTAL (BATCH * S_LEN)   // 8192

typedef __attribute__((ext_vector_type(16))) __bf16 v16bf;
typedef __attribute__((ext_vector_type(8)))  float  v8f;

static __device__ __forceinline__ __bf16 f2bf(float f) { return (__bf16)f; }

// Async global->LDS copy (CDNA5 GLOBAL_LOAD_ASYNC_TO_LDS_B128, GV mode).
// The low 32 bits of a generic pointer to LDS are the wave-relative LDS
// address (flat aperture truncation, ISA 10.2), which is what VDST wants.
static __device__ __forceinline__ void async_ld_b128(void* lds, const void* gaddr) {
    unsigned lds_off = (unsigned)(size_t)lds;
    asm volatile("global_load_async_to_lds_b128 %0, %1, off"
                 :: "v"(lds_off), "v"(gaddr)
                 : "memory");
}

// ---------------------------------------------------------------------------
// Kernel 1: QKV projection.  out = X @ W^T + b, converted to bf16.
// One wave computes one 16x16 tile of one projection (proj = 0/1/2 -> Q/K/V).
// Q,K stored [B,H,S,Dh] row-major; V stored transposed [B,H,Dh,S] so the
// attention PV matmul gets contiguous B-fragment loads.
// ---------------------------------------------------------------------------
__global__ __launch_bounds__(256)
void qkv_proj_kernel(const float* __restrict__ Xq,
                     const float* __restrict__ Xk,
                     const float* __restrict__ Xv,
                     const float* __restrict__ Wq, const float* __restrict__ bq,
                     const float* __restrict__ Wk, const float* __restrict__ bk,
                     const float* __restrict__ Wv, const float* __restrict__ bv,
                     __bf16* __restrict__ qbuf,
                     __bf16* __restrict__ kbuf,
                     __bf16* __restrict__ vTbuf)
{
    const int wave_global = (int)((blockIdx.x * blockDim.x + threadIdx.x) >> 5);
    const int lane = (int)(threadIdx.x & 31);
    const int hi   = lane >> 4;     // half-wave: 0 or 1
    const int l16  = lane & 15;

    const int ntiles = D_MODEL / 16;              // 32
    const int mtiles = M_TOTAL / 16;              // 512
    const int tiles_per_proj = ntiles * mtiles;   // 16384

    const int proj = wave_global / tiles_per_proj;
    const int rem  = wave_global % tiles_per_proj;
    const int mt   = rem / ntiles;
    const int nt   = rem % ntiles;

    const float* X    = (proj == 0) ? Xq : (proj == 1) ? Xk : Xv;
    const float* W    = (proj == 0) ? Wq : (proj == 1) ? Wk : Wv;
    const float* bias = (proj == 0) ? bq : (proj == 1) ? bk : bv;

    const int m0 = mt * 16;
    const int n0 = nt * 16;

    const float* Xrow = X + (size_t)(m0 + l16) * D_MODEL;
    const float* Wrow = W + (size_t)(n0 + l16) * D_MODEL;

    const int ka = hi * 16;   // A-frag K base inside 32-chunk (contiguous, same as B)
    const int kb = hi * 16;   // B-frag K base inside 32-chunk (contiguous, same as A)

    v8f acc = {};
    for (int k0 = 0; k0 < D_MODEL; k0 += 32) {
        v16bf a, b;
#pragma unroll
        for (int e = 0; e < 16; ++e)
            a[e] = f2bf(Xrow[k0 + ka + e]);
#pragma unroll
        for (int e = 0; e < 16; ++e)
            b[e] = f2bf(Wrow[k0 + kb + e]);
        acc = __builtin_amdgcn_wmma_f32_16x16x32_bf16(
                  false, a, false, b, (short)0, acc, false, false);
    }

    const float bv_ = bias[n0 + l16];
    const int o  = n0 + l16;          // output feature
    const int h  = o >> 6;            // head
    const int d  = o & 63;            // dim within head

#pragma unroll
    for (int r = 0; r < 8; ++r) {
        const float val = acc[r] + bv_;
        const int gm = m0 + r + 8 * hi;     // global row (b*S + s)
        const int bi = gm >> 12;
        const int s  = gm & (S_LEN - 1);
        if (proj == 0) {
            qbuf[((((size_t)bi * N_HEADS + h) * S_LEN) + s) * HEAD_DIM + d] = f2bf(val);
        } else if (proj == 1) {
            kbuf[((((size_t)bi * N_HEADS + h) * S_LEN) + s) * HEAD_DIM + d] = f2bf(val);
        } else {
            vTbuf[(((size_t)bi * N_HEADS + h) * HEAD_DIM + d) * S_LEN + s] = f2bf(val);
        }
    }
}

// ---------------------------------------------------------------------------
// Kernel 2: block-cooperative flash attention.
// One block (8 waves) owns 128 query rows of one (b,h).  K / V^T tiles for 32
// keys are staged into LDS with async b128 loads (double-buffered, ASYNCcnt)
// and shared by all 8 waves -> 8x less L2 traffic than per-wave streaming.
// ---------------------------------------------------------------------------
__global__ __launch_bounds__(256)
void attn_kernel(const __bf16* __restrict__ qbuf,
                 const __bf16* __restrict__ kbuf,
                 const __bf16* __restrict__ vTbuf,
                 const float*  __restrict__ mask,
                 float* __restrict__ out)
{
    __shared__ __bf16 klds[2][32 * HEAD_DIM];   // 2 x 4 KB: 32 key rows x 64
    __shared__ __bf16 vlds[2][HEAD_DIM * 32];   // 2 x 4 KB: 64 dims x 32 keys
    __shared__ __bf16 plds[8][16 * 32];         // per-wave P staging, 8 KB

    const int tid   = (int)threadIdx.x;
    const int wslot = tid >> 5;
    const int lane  = tid & 31;
    const int hi    = lane >> 4;
    const int l16   = lane & 15;

    // block -> (b,h) and 128-row query slab; wave -> one 16-row q tile
    const int bh     = (int)blockIdx.x >> 5;    // 0..15
    const int qblock = (int)blockIdx.x & 31;    // 0..31
    const int b  = bh >> 3;
    const int h  = bh & 7;
    const int q0 = (qblock * 8 + wslot) * 16;

    const __bf16* Q  = qbuf  + (size_t)bh * S_LEN * HEAD_DIM;
    const __bf16* K  = kbuf  + (size_t)bh * S_LEN * HEAD_DIM;
    const __bf16* VT = vTbuf + (size_t)bh * HEAD_DIM * S_LEN;
    const float*  Mb = mask  + (size_t)b * S_LEN * S_LEN;

    // cooperative-load coordinates (one b128 = 8 bf16 per thread per tile)
    const int krow = tid >> 3, kseg = tid & 7;  // 32 rows x 8 segs of K tile
    const int vrow = tid >> 2, vseg = tid & 3;  // 64 rows x 4 segs of V^T tile

    // Preload Q A-fragments for the whole scan: 16x64 -> two 16x32 frags.
    v16bf qa[2];
    {
        const __bf16* Qrow = Q + (size_t)(q0 + l16) * HEAD_DIM;
#pragma unroll
        for (int c = 0; c < 2; ++c) {
            const int kb0 = c * 32 + hi * 16;
#pragma unroll
            for (int e = 0; e < 16; ++e)
                qa[c][e] = Qrow[kb0 + e];
        }
    }

    v8f o0 = {}, o1 = {}, o2 = {}, o3 = {};
    float mrow[8], lrow[8];
#pragma unroll
    for (int r = 0; r < 8; ++r) { mrow[r] = -1e30f; lrow[r] = 0.0f; }

    const float inv_scale = 0.125f;          // 1/sqrt(64)
    __bf16* P = &plds[wslot][0];

    const int NIT = S_LEN / 32;              // 128

    // prologue: stage tile 0 into buffer 0
    async_ld_b128(&klds[0][krow * HEAD_DIM + kseg * 8], K + (size_t)krow * HEAD_DIM + kseg * 8);
    async_ld_b128(&vlds[0][vrow * 32 + vseg * 8],       VT + (size_t)vrow * S_LEN + vseg * 8);

    for (int it = 0; it < NIT; ++it) {
        const int k0  = it * 32;
        const int buf = it & 1;

        if (it + 1 < NIT) {   // prefetch next tile into the other buffer
            const int kn = k0 + 32;
            async_ld_b128(&klds[buf ^ 1][krow * HEAD_DIM + kseg * 8],
                          K + (size_t)(kn + krow) * HEAD_DIM + kseg * 8);
            async_ld_b128(&vlds[buf ^ 1][vrow * 32 + vseg * 8],
                          VT + (size_t)vrow * S_LEN + kn + vseg * 8);
            asm volatile("s_wait_asynccnt 0x2" ::: "memory");  // current tile landed
        } else {
            asm volatile("s_wait_asynccnt 0x0" ::: "memory");
        }
        __syncthreads();      // all threads' segments visible

        // ---- raw scores for 32 keys: two 16x16 tiles, Dh=64 in 2 K-steps ----
        v8f sc[2];
#pragma unroll
        for (int t = 0; t < 2; ++t) {
            v8f c = {};
#pragma unroll
            for (int dstep = 0; dstep < 2; ++dstep) {
                const __bf16* Krow = &klds[buf][(t * 16 + l16) * HEAD_DIM + dstep * 32 + hi * 16];
                v16bf kv;
#pragma unroll
                for (int e = 0; e < 16; ++e) kv[e] = Krow[e];
                c = __builtin_amdgcn_wmma_f32_16x16x32_bf16(
                        false, qa[dstep], false, kv, (short)0, c, false, false);
            }
            sc[t] = c;
        }

        // ---- logits = (score + mask) * inv_scale; online softmax (fp32) ----
        float pmax[8];
#pragma unroll
        for (int r = 0; r < 8; ++r) {
            const int qr = q0 + r + 8 * hi;
            const float mv0 = Mb[(size_t)qr * S_LEN + k0 + l16];
            const float mv1 = Mb[(size_t)qr * S_LEN + k0 + 16 + l16];
            sc[0][r] = (sc[0][r] + mv0) * inv_scale;
            sc[1][r] = (sc[1][r] + mv1) * inv_scale;
            pmax[r]  = fmaxf(sc[0][r], sc[1][r]);
        }
#pragma unroll
        for (int off = 8; off >= 1; off >>= 1)
#pragma unroll
            for (int r = 0; r < 8; ++r)
                pmax[r] = fmaxf(pmax[r], __shfl_xor(pmax[r], off, 32));

        float alpha[8], rsum[8];
#pragma unroll
        for (int r = 0; r < 8; ++r) {
            const float mnew = fmaxf(mrow[r], pmax[r]);
            alpha[r] = __expf(mrow[r] - mnew);
            mrow[r]  = mnew;
            const float p0 = __expf(sc[0][r] - mnew);
            const float p1 = __expf(sc[1][r] - mnew);
            sc[0][r] = p0; sc[1][r] = p1;
            rsum[r]  = p0 + p1;
        }
#pragma unroll
        for (int off = 8; off >= 1; off >>= 1)
#pragma unroll
            for (int r = 0; r < 8; ++r)
                rsum[r] += __shfl_xor(rsum[r], off, 32);
#pragma unroll
        for (int r = 0; r < 8; ++r) {
            lrow[r] = lrow[r] * alpha[r] + rsum[r];
            o0[r] *= alpha[r]; o1[r] *= alpha[r];
            o2[r] *= alpha[r]; o3[r] *= alpha[r];
        }

        // ---- C-layout P -> row-major LDS (private slice) -> A-layout frag ----
#pragma unroll
        for (int r = 0; r < 8; ++r) {
            const int row = r + 8 * hi;
            P[row * 32 + l16]      = f2bf(sc[0][r]);
            P[row * 32 + 16 + l16] = f2bf(sc[1][r]);
        }
        asm volatile("s_wait_dscnt 0x0" ::: "memory");   // same-wave RAW on slice

        v16bf pa;
        {
            const __bf16* prow = P + l16 * 32;
            const int kb0 = hi * 16;
#pragma unroll
            for (int e = 0; e < 16; ++e)
                pa[e] = prow[kb0 + e];
        }

        // ---- O += P @ V : 4 d-tiles of 16, K=32 keys ----
#pragma unroll
        for (int t = 0; t < 4; ++t) {
            const __bf16* vr = &vlds[buf][(t * 16 + l16) * 32 + hi * 16];
            v16bf vb;
#pragma unroll
            for (int e = 0; e < 16; ++e) vb[e] = vr[e];
            v8f* op = (t == 0) ? &o0 : (t == 1) ? &o1 : (t == 2) ? &o2 : &o3;
            *op = __builtin_amdgcn_wmma_f32_16x16x32_bf16(
                      false, pa, false, vb, (short)0, *op, false, false);
        }

        __syncthreads();      // tile fully consumed before it can be re-staged
    }

    // ---- normalize and write out[b, s, h*64 + d] ----
#pragma unroll
    for (int r = 0; r < 8; ++r) {
        const float inv_l = 1.0f / lrow[r];
        const int srow = q0 + r + 8 * hi;
        float* dst = out + ((size_t)b * S_LEN + srow) * D_MODEL + h * HEAD_DIM;
        dst[0 * 16 + l16] = o0[r] * inv_l;
        dst[1 * 16 + l16] = o1[r] * inv_l;
        dst[2 * 16 + l16] = o2[r] * inv_l;
        dst[3 * 16 + l16] = o3[r] * inv_l;
    }
}

// ---------------------------------------------------------------------------
extern "C" void kernel_launch(void* const* d_in, const int* in_sizes, int n_in,
                              void* d_out, int out_size, void* d_ws, size_t ws_size,
                              hipStream_t stream) {
    const float* query = (const float*)d_in[0];
    const float* key_t = (const float*)d_in[1];
    const float* value = (const float*)d_in[2];
    const float* mask  = (const float*)d_in[3];
    const float* Wq    = (const float*)d_in[4];
    const float* bq    = (const float*)d_in[5];
    const float* Wk    = (const float*)d_in[6];
    const float* bk    = (const float*)d_in[7];
    const float* Wv    = (const float*)d_in[8];
    const float* bv    = (const float*)d_in[9];
    float* out = (float*)d_out;

    const size_t elems = (size_t)BATCH * N_HEADS * S_LEN * HEAD_DIM;  // 4M
    __bf16* qbuf = (__bf16*)d_ws;
    __bf16* kbuf = qbuf + elems;
    __bf16* vT   = kbuf + elems;   // total 24 MB of workspace

    dim3 blk(256);

    const int proj_waves = 3 * (M_TOTAL / 16) * (D_MODEL / 16);   // 49152
    qkv_proj_kernel<<<dim3(proj_waves / 8), blk, 0, stream>>>(
        query, key_t, value, Wq, bq, Wk, bk, Wv, bv, qbuf, kbuf, vT);

    // one block per (b,h, 128-query slab): 2*8*32 = 512 blocks of 8 waves
    attn_kernel<<<dim3(BATCH * N_HEADS * (S_LEN / 128)), blk, 0, stream>>>(
        qbuf, kbuf, vT, mask, out);
}
